// GRU_cell_my_50809463111717
// MI455X (gfx1250) — matrix operand order repaired and checked by reading, not compiled, not run on hardware
//
#include <hip/hip_runtime.h>

// ---------------------------------------------------------------------------
// GRU on MI455X (gfx1250, wave32, WMMA).
//
//  * All operands converted to bf16 once; weights pre-transposed (N-major) so
//    WMMA B fragments are two contiguous dwordx4 loads.
//  * Time-parallel input projection X@[Wxz|Wxr|Wxh] hoisted into one big
//    bf16 WMMA GEMM (32768x1024 @ 1024x6144); 8 waves/block cooperate on a
//    128x64 macro-tile with the B tile staged in LDS via
//    global_load_async_to_lds_b128 (8x less B traffic).
//  * Serial recurrence: per step, waves own ALL of M (64 rows) x 16 cols so
//    every recurrent-weight byte is read exactly once per step from the
//    L2-resident (192MB) bf16 weights (~28MB/step -> ~14GB over 512 steps).
//  * fp32 accumulation; sigmoid/tanh epilogues in fp32.
//
// Workspace: ~515 MB (dominated by bf16 Xproj, 32768x6144x2B).
// ---------------------------------------------------------------------------

#define T_STEPS 512
#define BATCH   64
#define IN_DIM  1024
#define HID     2048
#define N3      (3 * HID)   // 6144

typedef unsigned short bf16_t;
typedef __attribute__((ext_vector_type(16))) __bf16 v16bf;
typedef __attribute__((ext_vector_type(8)))  float  v8f;
typedef int i4v __attribute__((vector_size(16)));   // matches builtin pointee

union FragU { uint4 q[2]; v16bf v; };

#if defined(__has_builtin)
#if __has_builtin(__builtin_amdgcn_global_load_async_to_lds_b128)
#define HAVE_ASYNC_LDS 1
#endif
#endif

__device__ __forceinline__ bf16_t f2bf(float f) {
    unsigned u = __float_as_uint(f);
    return (bf16_t)((u + 0x7FFFu + ((u >> 16) & 1u)) >> 16);   // RNE
}
__device__ __forceinline__ float bf2f(bf16_t h) {
    return __uint_as_float(((unsigned)h) << 16);
}
__device__ __forceinline__ float sigmoidf_(float x) {
    return 1.0f / (1.0f + __expf(-x));
}

// A fragment: 16x32 bf16, row-major source (ISA 7.12.2).
// lane L: m = L&15 ; v0..7 hold K = k0+16*(L>>4)+0..15 (contiguous).
__device__ __forceinline__ v16bf load_frag_A(const bf16_t* __restrict__ A,
                                             int lda, int m0, int k0, int lane) {
    const int half = lane >> 4, mr = lane & 15;
    const bf16_t* p = A + (size_t)(m0 + mr) * lda + k0 + 16 * half;
    FragU f;
    f.q[0] = *(const uint4*)(p);
    f.q[1] = *(const uint4*)(p + 8);
    return f.v;
}

// B fragment: 32x16 bf16 from an N-major (transposed) weight, ldb = K.
// lane L: n = L&15 ; v0..7 hold K = k0+16*(L>>4)+0..15 (contiguous).
__device__ __forceinline__ v16bf load_frag_B(const bf16_t* __restrict__ Bt,
                                             int ldb, int n0, int k0, int lane) {
    const int half = lane >> 4, nc = lane & 15;
    const bf16_t* p = Bt + (size_t)(n0 + nc) * ldb + k0 + 16 * half;
    FragU f;
    f.q[0] = *(const uint4*)(p);
    f.q[1] = *(const uint4*)(p + 8);
    return f.v;
}

// B fragment out of an LDS-staged 64x32 tile, row stride LDSB_LD shorts.
#define LDSB_LD 40   // 80B rows: 16B aligned, spreads banks
__device__ __forceinline__ v16bf load_frag_B_lds(const bf16_t* ldsB,
                                                 int nloc0, int lane) {
    const int half = lane >> 4, nc = lane & 15;
    const bf16_t* p = ldsB + (nloc0 + nc) * LDSB_LD + 16 * half;
    FragU f;
    f.q[0] = *(const uint4*)(p);
    f.q[1] = *(const uint4*)(p + 8);
    return f.v;
}

__device__ __forceinline__ v8f wmma_bf16(v16bf a, v16bf b, v8f c) {
    return __builtin_amdgcn_wmma_f32_16x16x32_bf16(
        false, a, false, b, (short)0, c, false, false);
}

// --------------------------- conversion kernels -----------------------------

__global__ void k_f2bf(const float* __restrict__ src, bf16_t* __restrict__ dst,
                       long n) {
    long i = (long)blockIdx.x * blockDim.x + threadIdx.x;
    if (i < n) dst[i] = f2bf(src[i]);
}

__global__ void k_copyf(const float* __restrict__ src, float* __restrict__ dst,
                        long n) {
    long i = (long)blockIdx.x * blockDim.x + threadIdx.x;
    if (i < n) dst[i] = src[i];
}

// dst[n*K + k] = bf16(src[k*N + n])   (K x N fp32 -> N x K bf16)
__global__ void k_trcvt(const float* __restrict__ src, bf16_t* __restrict__ dst,
                        int K, int N) {
    long i = (long)blockIdx.x * blockDim.x + threadIdx.x;
    long total = (long)K * N;
    if (i < total) {
        int n = (int)(i % N);
        int k = (int)(i / N);
        dst[(size_t)n * K + k] = f2bf(src[i]);
    }
}

// ------------------------ big time-parallel projection ----------------------
// 8 waves/block compute a 128(M) x 64(N) macro tile; the 64x32 bf16 B tile is
// staged in LDS once per K-chunk and shared by all waves (8x B-traffic cut).
__global__ void __launch_bounds__(256)
k_xproj(const bf16_t* __restrict__ Xbf,
        const bf16_t* __restrict__ Wxt,
        const float* __restrict__ bz,
        const float* __restrict__ br,
        const float* __restrict__ bh,
        bf16_t* __restrict__ Xproj) {
    __shared__ bf16_t ldsB[64 * LDSB_LD];

    const int lane = threadIdx.x & 31;
    const int wv   = threadIdx.x >> 5;          // 0..7
    const int NSTRIP = N3 / 64;                 // 96
    const int bm = blockIdx.x / NSTRIP;         // 0..255
    const int ns = blockIdx.x % NSTRIP;
    const int m0 = bm * 128 + wv * 16;          // this wave's 16 rows
    const int n0 = ns * 64;

    // staging assignment: 64 rows x 4 chunks of 16B
    const int srow = threadIdx.x >> 2;          // 0..63
    const int schk = (threadIdx.x & 3) * 8;     // 0,8,16,24 shorts

    v8f acc[4] = {};
    for (int k0 = 0; k0 < IN_DIM; k0 += 32) {
        __syncthreads();   // previous chunk fully consumed
        {
            const bf16_t* gp = Wxt + (size_t)(n0 + srow) * IN_DIM + k0 + schk;
            bf16_t* lp = ldsB + srow * LDSB_LD + schk;
#ifdef HAVE_ASYNC_LDS
            __builtin_amdgcn_global_load_async_to_lds_b128(
                (__attribute__((address_space(1))) i4v*)(i4v*)gp,
                (__attribute__((address_space(3))) i4v*)(i4v*)lp, 0, 0);
#if __has_builtin(__builtin_amdgcn_s_wait_asynccnt)
            __builtin_amdgcn_s_wait_asynccnt(0);
#else
            asm volatile("s_wait_asynccnt 0x0" ::: "memory");
#endif
#else
            *(uint4*)lp = *(const uint4*)gp;    // sync fallback (ds_store_b128)
#endif
        }
        __syncthreads();   // B tile visible to all waves

        v16bf a = load_frag_A(Xbf, IN_DIM, m0, k0, lane);
        if (k0 + 32 < IN_DIM)  // software prefetch of next A chunk
            __builtin_prefetch(Xbf + (size_t)(m0 + (lane & 15)) * IN_DIM + k0 + 32, 0, 0);
#pragma unroll
        for (int nt = 0; nt < 4; ++nt) {
            v16bf b = load_frag_B_lds(ldsB, nt * 16, lane);
            acc[nt] = wmma_bf16(a, b, acc[nt]);
        }
    }

    const int half = lane >> 4, nc = lane & 15;
    const int plane = n0 / HID;                 // uniform per block
    const float* bias = (plane == 0) ? bz : (plane == 1) ? br : bh;
#pragma unroll
    for (int nt = 0; nt < 4; ++nt) {
        int n = n0 + nt * 16 + nc;
        float bv = bias[n % HID];
#pragma unroll
        for (int j = 0; j < 8; ++j) {
            int m = m0 + j + 8 * half;
            Xproj[(size_t)m * N3 + n] = f2bf(acc[nt][j] + bv);
        }
    }
}

// ------------------------------ recurrence ----------------------------------
// Per-step kernels: each wave owns ALL of M (64 rows) x one 16-col N tile, so
// each weight byte is loaded exactly once per step (minimal L2 traffic on the
// serial critical path); B fragments are reused across the 4 M-tiles.

// Step kernel 1: Z = sigmoid(xz + H@Whz), R = sigmoid(xr + H@Whr), RH = R*H.
__global__ void __launch_bounds__(256)
k_gates(int t,
        const bf16_t* __restrict__ Hbf,
        const bf16_t* __restrict__ Whzt,
        const bf16_t* __restrict__ Whrt,
        const bf16_t* __restrict__ Xproj,
        const float* __restrict__ H,
        float* __restrict__ Z,
        bf16_t* __restrict__ RHbf) {
    const int lane = threadIdx.x & 31;
    const int wave = blockIdx.x * (blockDim.x >> 5) + (threadIdx.x >> 5);
    const int n0 = wave * 16;                   // 0..2032

    v8f az[4] = {}, ar[4] = {};
    for (int k0 = 0; k0 < HID; k0 += 32) {
        v16bf bz_ = load_frag_B(Whzt, HID, n0, k0, lane);
        v16bf br_ = load_frag_B(Whrt, HID, n0, k0, lane);
#pragma unroll
        for (int mt = 0; mt < 4; ++mt) {
            v16bf a = load_frag_A(Hbf, HID, mt * 16, k0, lane);
            az[mt] = wmma_bf16(a, bz_, az[mt]);
            ar[mt] = wmma_bf16(a, br_, ar[mt]);
        }
    }

    const int half = lane >> 4, nc = lane & 15;
    const int n = n0 + nc;
#pragma unroll
    for (int mt = 0; mt < 4; ++mt) {
#pragma unroll
        for (int j = 0; j < 8; ++j) {
            int m = mt * 16 + j + 8 * half;
            size_t xi = (size_t)(t * BATCH + m) * N3 + n;
            float z = sigmoidf_(az[mt][j] + bf2f(Xproj[xi]));
            float r = sigmoidf_(ar[mt][j] + bf2f(Xproj[xi + HID]));
            float h = H[(size_t)m * HID + n];
            Z[(size_t)m * HID + n] = z;
            RHbf[(size_t)m * HID + n] = f2bf(r * h);
        }
    }
}

// Step kernel 2: Htilda = tanh(xh + RH@Whh); Hnew = Z*H + (1-Z)*Htilda.
__global__ void __launch_bounds__(256)
k_cand(int t,
       const bf16_t* __restrict__ RHbf,
       const bf16_t* __restrict__ Whht,
       const bf16_t* __restrict__ Xproj,
       const float* __restrict__ Z,
       float* __restrict__ H,
       bf16_t* __restrict__ Hbf) {
    const int lane = threadIdx.x & 31;
    const int wave = blockIdx.x * (blockDim.x >> 5) + (threadIdx.x >> 5);
    const int n0 = wave * 16;

    v8f acc[4] = {};
    for (int k0 = 0; k0 < HID; k0 += 32) {
        v16bf b = load_frag_B(Whht, HID, n0, k0, lane);
#pragma unroll
        for (int mt = 0; mt < 4; ++mt) {
            v16bf a = load_frag_A(RHbf, HID, mt * 16, k0, lane);
            acc[mt] = wmma_bf16(a, b, acc[mt]);
        }
    }

    const int half = lane >> 4, nc = lane & 15;
    const int n = n0 + nc;
#pragma unroll
    for (int mt = 0; mt < 4; ++mt) {
#pragma unroll
        for (int j = 0; j < 8; ++j) {
            int m = mt * 16 + j + 8 * half;
            size_t xi = (size_t)(t * BATCH + m) * N3 + n;
            float ht = tanhf(acc[mt][j] + bf2f(Xproj[xi + 2 * HID]));
            size_t hi = (size_t)m * HID + n;
            float z = Z[hi];
            float hn = z * H[hi] + (1.0f - z) * ht;
            H[hi]   = hn;                // per-element read-then-write: safe
            Hbf[hi] = f2bf(hn);
        }
    }
}

// Step kernel 3: Y_t = Hnew @ Whq + b_hq  -> d_out[t*B*I : ...]
__global__ void __launch_bounds__(256)
k_out(int t,
      const bf16_t* __restrict__ Hbf,
      const bf16_t* __restrict__ Whqt,
      const float* __restrict__ bhq,
      float* __restrict__ out) {
    const int lane = threadIdx.x & 31;
    const int wave = blockIdx.x * (blockDim.x >> 5) + (threadIdx.x >> 5);
    const int n0 = wave * 16;                   // 0..1008

    v8f acc[4] = {};
    for (int k0 = 0; k0 < HID; k0 += 32) {
        v16bf b = load_frag_B(Whqt, HID, n0, k0, lane);
#pragma unroll
        for (int mt = 0; mt < 4; ++mt) {
            v16bf a = load_frag_A(Hbf, HID, mt * 16, k0, lane);
            acc[mt] = wmma_bf16(a, b, acc[mt]);
        }
    }

    const int half = lane >> 4, nc = lane & 15;
    const int n = n0 + nc;
    float bv = bhq[n];
    float* dst = out + (size_t)t * BATCH * IN_DIM;
#pragma unroll
    for (int mt = 0; mt < 4; ++mt) {
#pragma unroll
        for (int j = 0; j < 8; ++j) {
            int m = mt * 16 + j + 8 * half;
            dst[(size_t)m * IN_DIM + n] = acc[mt][j] + bv;
        }
    }
}

// ------------------------------- launcher -----------------------------------

extern "C" void kernel_launch(void* const* d_in, const int* in_sizes, int n_in,
                              void* d_out, int out_size, void* d_ws, size_t ws_size,
                              hipStream_t stream) {
    (void)in_sizes; (void)n_in; (void)out_size; (void)ws_size;

    const float* inputs = (const float*)d_in[0];   // (T,B,I)
    const float* state  = (const float*)d_in[1];   // (B,Hd)
    const float* W_xz   = (const float*)d_in[2];
    const float* W_hz   = (const float*)d_in[3];
    const float* b_z    = (const float*)d_in[4];
    const float* W_xr   = (const float*)d_in[5];
    const float* W_hr   = (const float*)d_in[6];
    const float* b_r    = (const float*)d_in[7];
    const float* W_xh   = (const float*)d_in[8];
    const float* W_hh   = (const float*)d_in[9];
    const float* b_h    = (const float*)d_in[10];
    const float* W_hq   = (const float*)d_in[11];
    const float* b_hq   = (const float*)d_in[12];
    float* out = (float*)d_out;

    // --- workspace carve-out (256B aligned) ---
    char* ws = (char*)d_ws;
    size_t off = 0;
    auto carve = [&](size_t bytes) -> void* {
        void* p = ws + off;
        off += (bytes + 255) & ~(size_t)255;
        return p;
    };
    const long nX = (long)T_STEPS * BATCH * IN_DIM;          // 33,554,432
    bf16_t* Xbf   = (bf16_t*)carve((size_t)nX * 2);
    bf16_t* Wxt   = (bf16_t*)carve((size_t)N3 * IN_DIM * 2); // 6144x1024
    bf16_t* Whzt  = (bf16_t*)carve((size_t)HID * HID * 2);
    bf16_t* Whrt  = (bf16_t*)carve((size_t)HID * HID * 2);
    bf16_t* Whht  = (bf16_t*)carve((size_t)HID * HID * 2);
    bf16_t* Whqt  = (bf16_t*)carve((size_t)IN_DIM * HID * 2);
    bf16_t* Xproj = (bf16_t*)carve((size_t)T_STEPS * BATCH * N3 * 2);
    float*  H     = (float*) carve((size_t)BATCH * HID * 4);
    float*  Z     = (float*) carve((size_t)BATCH * HID * 4);
    bf16_t* Hbf   = (bf16_t*)carve((size_t)BATCH * HID * 2);
    bf16_t* RHbf  = (bf16_t*)carve((size_t)BATCH * HID * 2);

    // --- one-time conversions (bf16 + weight transposes) ---
    k_f2bf<<<(unsigned)((nX + 255) / 256), 256, 0, stream>>>(inputs, Xbf, nX);
    {
        long nw = (long)IN_DIM * HID;   // 2,097,152
        unsigned g = (unsigned)((nw + 255) / 256);
        k_trcvt<<<g, 256, 0, stream>>>(W_xz, Wxt + (size_t)0 * HID * IN_DIM, IN_DIM, HID);
        k_trcvt<<<g, 256, 0, stream>>>(W_xr, Wxt + (size_t)1 * HID * IN_DIM, IN_DIM, HID);
        k_trcvt<<<g, 256, 0, stream>>>(W_xh, Wxt + (size_t)2 * HID * IN_DIM, IN_DIM, HID);
        long nh = (long)HID * HID;      // 4,194,304
        unsigned gh = (unsigned)((nh + 255) / 256);
        k_trcvt<<<gh, 256, 0, stream>>>(W_hz, Whzt, HID, HID);
        k_trcvt<<<gh, 256, 0, stream>>>(W_hr, Whrt, HID, HID);
        k_trcvt<<<gh, 256, 0, stream>>>(W_hh, Whht, HID, HID);
        k_trcvt<<<g,  256, 0, stream>>>(W_hq, Whqt, HID, IN_DIM);
    }
    {
        long nh = (long)BATCH * HID;    // 131,072
        k_f2bf <<<(unsigned)((nh + 255) / 256), 256, 0, stream>>>(state, Hbf, nh);
        k_copyf<<<(unsigned)((nh + 255) / 256), 256, 0, stream>>>(state, H, nh);
    }

    // --- time-parallel input projection: 128x64 macro tiles, LDS-shared B ---
    {
        unsigned blocks = ((unsigned)(T_STEPS * BATCH) / 128) * (N3 / 64); // 24576
        k_xproj<<<blocks, 256, 0, stream>>>(Xbf, Wxt, b_z, b_r, b_h, Xproj);
    }

    // --- sequential recurrence: 128 / 128 / 64 waves per step ---
    for (int t = 0; t < T_STEPS; ++t) {
        k_gates<<<16, 256, 0, stream>>>(t, Hbf, Whzt, Whrt, Xproj, H, Z, RHbf);
        k_cand <<<16, 256, 0, stream>>>(t, RHbf, Whht, Xproj, Z, H, Hbf);
        k_out  <<< 8, 256, 0, stream>>>(t, Hbf, Whqt, b_hq, out);
    }

    // --- H_final appended after the (T*B, I) outputs ---
    k_copyf<<<512, 256, 0, stream>>>(H, out + (size_t)nX, (long)BATCH * HID);
}
